// TransformerEncoderBlock_80874234183908
// MI455X (gfx1250) — matrix operand order repaired and checked by reading, not compiled, not run on hardware
//
#include <hip/hip_runtime.h>
#include <hip/hip_bf16.h>

// ---------------------------------------------------------------------------
// Transformer encoder block on gfx1250 (MI455X), bf16 WMMA everywhere.
// B=2, S=2048, D=768, H=12, DK=DV=64, DFF=3072. N = B*S = 4096 rows.
// Async global->LDS staging (ASYNCcnt) + double-buffered GEMM pipeline.
// ---------------------------------------------------------------------------

typedef __bf16 bf16;
typedef __attribute__((ext_vector_type(16))) __bf16 v16bf;
typedef __attribute__((ext_vector_type(8)))  __bf16 v8bf;
typedef __attribute__((ext_vector_type(8)))  float  v8f;

#define CB 2
#define CS 2048
#define CD 768
#define CH 12
#define CDK 64
#define CDFF 3072
#define CN (CB * CS)

static __device__ __forceinline__ v16bf mk16(v8bf lo, v8bf hi) {
  v16bf r;
#pragma unroll
  for (int i = 0; i < 8; ++i) { r[i] = lo[i]; r[i + 8] = hi[i]; }
  return r;
}

// A-matrix 16x32 bf16 fragment: lane half selects base=0/8; per-lane elements
// are row[base+0..7] and row[base+16..23] (ISA 7.12.2, 16-bit A layout).
static __device__ __forceinline__ v16bf ld_afrag(const bf16* rowp, int base) {
  v8bf lo = *(const v8bf*)(rowp + base);
  v8bf hi = *(const v8bf*)(rowp + base + 16);
  return mk16(lo, hi);
}

// B-matrix 32x16 bf16 fragment (stored transposed as Bt[n][k]): lane half
// selects base=0/8; elements are K base+0..7 and base+16..23 (matches A).
static __device__ __forceinline__ v16bf ld_bfrag(const bf16* rowp, int lhalf) {
  v8bf lo = *(const v8bf*)(rowp + lhalf * 8);
  v8bf hi = *(const v8bf*)(rowp + lhalf * 8 + 16);
  return mk16(lo, hi);
}

static __device__ __forceinline__ v8f wmma_bf16(v16bf a, v16bf b, v8f c) {
  return __builtin_amdgcn_wmma_f32_16x16x32_bf16(false, a, false, b,
                                                 (short)0, c, false, false);
}

// Async 16B global -> LDS copy (GLOBAL_LOAD_ASYNC_TO_LDS_B128, ASYNCcnt).
static __device__ __forceinline__ void async_cp16(void* lds, const void* gptr) {
  unsigned l = (unsigned)(size_t)lds;  // generic addr [31:0] == LDS offset
  asm volatile("global_load_async_to_lds_b128 %0, %1, off"
               :: "v"(l), "v"(gptr) : "memory");
}
static __device__ __forceinline__ void wait_async0() {
  asm volatile("s_wait_asynccnt 0x0" ::: "memory");
}

// ---------------------------------------------------------------------------
// elementwise f32 -> bf16 cast (4 elems / thread)
// ---------------------------------------------------------------------------
__global__ __launch_bounds__(256) void cast_f32_bf16_kernel(
    const float* __restrict__ in, bf16* __restrict__ out) {
  int i4 = (blockIdx.x * 256 + threadIdx.x) * 4;
  float4 f = *(const float4*)(in + i4);
  out[i4 + 0] = (bf16)f.x;
  out[i4 + 1] = (bf16)f.y;
  out[i4 + 2] = (bf16)f.z;
  out[i4 + 3] = (bf16)f.w;
}

// ---------------------------------------------------------------------------
// Generic bf16 GEMM: C[128x64] tile per block, K-loop in steps of 32,
// double-buffered LDS. A tile staged with async LDS loads; B weight tile
// (f32) loaded before the WMMA block and converted/stored after it so the
// global latency hides under the matrix ops. One barrier per k-step.
// ---------------------------------------------------------------------------
#define GLDA 40  // padded LDS strides (80B rows, 16B-aligned chunks)
#define GLDB 40

__global__ __launch_bounds__(256) void gemm_bf16_kernel(
    const bf16* __restrict__ A, int lda,
    const float* __restrict__ Bw, long long headStride, int ldb, int K,
    const float* __restrict__ bias, const float* __restrict__ resid,
    float* __restrict__ outF, bf16* __restrict__ outB, int ldo,
    int relu, int headed) {
  __shared__ bf16 As[2][128 * GLDA];
  __shared__ bf16 Bt[2][64 * GLDB];

  const int tid = threadIdx.x;
  const int wave = tid >> 5, lane = tid & 31;
  const int l16 = lane & 15, lhalf = lane >> 4, base = lhalf * 8;
  const int n0 = blockIdx.x * 128, col0 = blockIdx.y * 64;
  const float* Bh = Bw + (size_t)blockIdx.z * (size_t)headStride;

  // Per-thread staging coordinates.
  const int arow0 = tid >> 2, aoff0 = (tid & 3) * 8;          // A chunk 0
  const int arow1 = (tid + 256) >> 2, aoff1 = ((tid + 256) & 3) * 8;
  const int e0 = (tid * 2) * 4, e1 = (tid * 2 + 1) * 4;       // B elems
  const int bk0 = e0 >> 6, bn0 = e0 & 63;
  const int bk1 = e1 >> 6, bn1 = e1 & 63;

  v8f acc[4];
#pragma unroll
  for (int cs = 0; cs < 4; ++cs)
#pragma unroll
    for (int r = 0; r < 8; ++r) acc[cs][r] = 0.0f;

  const int NK = K >> 5;

  // ---- prologue: stage k-chunk 0 into buffer 0 ----
  async_cp16(&As[0][arow0 * GLDA + aoff0],
             A + (size_t)(n0 + arow0) * lda + aoff0);
  async_cp16(&As[0][arow1 * GLDA + aoff1],
             A + (size_t)(n0 + arow1) * lda + aoff1);
  {
    float4 f0 = *(const float4*)(Bh + (size_t)bk0 * ldb + col0 + bn0);
    float4 f1 = *(const float4*)(Bh + (size_t)bk1 * ldb + col0 + bn1);
    Bt[0][(bn0 + 0) * GLDB + bk0] = (bf16)f0.x;
    Bt[0][(bn0 + 1) * GLDB + bk0] = (bf16)f0.y;
    Bt[0][(bn0 + 2) * GLDB + bk0] = (bf16)f0.z;
    Bt[0][(bn0 + 3) * GLDB + bk0] = (bf16)f0.w;
    Bt[0][(bn1 + 0) * GLDB + bk1] = (bf16)f1.x;
    Bt[0][(bn1 + 1) * GLDB + bk1] = (bf16)f1.y;
    Bt[0][(bn1 + 2) * GLDB + bk1] = (bf16)f1.z;
    Bt[0][(bn1 + 3) * GLDB + bk1] = (bf16)f1.w;
  }
  wait_async0();
  __syncthreads();

  int cur = 0;
  for (int kc = 0; kc < NK; ++kc) {
    const int nxt = cur ^ 1;
    const int kn = (kc + 1) * 32;
    const bool pre = (kc + 1 < NK);
    float4 f0 = {0, 0, 0, 0}, f1 = {0, 0, 0, 0};
    if (pre) {
      // async A tile for next chunk (no wait yet)
      async_cp16(&As[nxt][arow0 * GLDA + aoff0],
                 A + (size_t)(n0 + arow0) * lda + kn + aoff0);
      async_cp16(&As[nxt][arow1 * GLDA + aoff1],
                 A + (size_t)(n0 + arow1) * lda + kn + aoff1);
      // B weight loads for next chunk (converted after compute)
      f0 = *(const float4*)(Bh + (size_t)(kn + bk0) * ldb + col0 + bn0);
      f1 = *(const float4*)(Bh + (size_t)(kn + bk1) * ldb + col0 + bn1);
      if (kc + 2 < NK)  // speculative pull of the tile after that
        __builtin_prefetch(Bh + (size_t)(kn + 32 + bk0) * ldb + col0 + bn0, 0, 1);
    }

    // ---- compute from buffer `cur` ----
    const int rowA = wave * 16 + l16;
    v16bf a = ld_afrag(&As[cur][rowA * GLDA], base);
    v16bf b0 = ld_bfrag(&Bt[cur][(0 * 16 + l16) * GLDB], lhalf);
    v16bf b1 = ld_bfrag(&Bt[cur][(1 * 16 + l16) * GLDB], lhalf);
    v16bf b2 = ld_bfrag(&Bt[cur][(2 * 16 + l16) * GLDB], lhalf);
    v16bf b3 = ld_bfrag(&Bt[cur][(3 * 16 + l16) * GLDB], lhalf);
    acc[0] = wmma_bf16(a, b0, acc[0]);
    acc[1] = wmma_bf16(a, b1, acc[1]);
    acc[2] = wmma_bf16(a, b2, acc[2]);
    acc[3] = wmma_bf16(a, b3, acc[3]);

    if (pre) {  // convert + store B tile for next chunk
      Bt[nxt][(bn0 + 0) * GLDB + bk0] = (bf16)f0.x;
      Bt[nxt][(bn0 + 1) * GLDB + bk0] = (bf16)f0.y;
      Bt[nxt][(bn0 + 2) * GLDB + bk0] = (bf16)f0.z;
      Bt[nxt][(bn0 + 3) * GLDB + bk0] = (bf16)f0.w;
      Bt[nxt][(bn1 + 0) * GLDB + bk1] = (bf16)f1.x;
      Bt[nxt][(bn1 + 1) * GLDB + bk1] = (bf16)f1.y;
      Bt[nxt][(bn1 + 2) * GLDB + bk1] = (bf16)f1.z;
      Bt[nxt][(bn1 + 3) * GLDB + bk1] = (bf16)f1.w;
    }
    wait_async0();     // async writes into nxt complete
    __syncthreads();   // everyone done reading cur / writing nxt
    cur = nxt;
  }

  // Epilogue. C layout: vgpr r, lane -> M = r + 8*lhalf, N = l16.
#pragma unroll
  for (int cs = 0; cs < 4; ++cs) {
#pragma unroll
    for (int r = 0; r < 8; ++r) {
      int n = n0 + wave * 16 + r + 8 * lhalf;
      int col = col0 + cs * 16 + l16;
      float v = acc[cs][r];
      if (bias) v += bias[col];
      if (resid) v += resid[(size_t)n * ldo + col];
      if (relu) v = fmaxf(v, 0.0f);
      size_t ro;
      if (headed) {
        int b = n >> 11, s = n & (CS - 1);
        ro = (size_t)(b * CH + (int)blockIdx.z) * CS + s;
      } else {
        ro = (size_t)n;
      }
      if (outF) outF[ro * ldo + col] = v;
      else      outB[ro * ldo + col] = (bf16)v;
    }
  }
}

// ---------------------------------------------------------------------------
// Attention pass A: column-softmax stats (softmax over the QUERY axis).
// Per key s: m[s]=max_q score, denom[s]=sum_q exp. Block = 64 keys of one
// (b,h); waves stream all 2048 queries with online (m,sum) update.
// ---------------------------------------------------------------------------
__global__ __launch_bounds__(256) void attn_stats_kernel(
    const bf16* __restrict__ Qb, const bf16* __restrict__ Kb,
    float* __restrict__ m_arr, float* __restrict__ r_arr) {
  __shared__ bf16 Ks[64 * 72];
  __shared__ float lm[8][64];
  __shared__ float ls[8][64];

  const int tid = threadIdx.x;
  const int wave = tid >> 5, lane = tid & 31;
  const int l16 = lane & 15, lhalf = lane >> 4, base = lhalf * 8;
  const int bh = blockIdx.z * CH + blockIdx.y;
  const int s0 = blockIdx.x * 64;
  const bf16* Kbase = Kb + (size_t)bh * CS * CDK;
  const bf16* Qbase = Qb + (size_t)bh * CS * CDK;

#pragma unroll
  for (int i = 0; i < 2; ++i) {  // K tile 64x64 bf16 -> LDS (async)
    int c = tid + i * 256;
    int row = c >> 3, off = (c & 7) * 8;
    async_cp16(&Ks[row * 72 + off],
               Kbase + (size_t)(s0 + row) * CDK + off);
  }
  wait_async0();
  __syncthreads();

  float mst[4], sst[4];
#pragma unroll
  for (int cs = 0; cs < 4; ++cs) { mst[cs] = -3.0e38f; sst[cs] = 0.0f; }

  for (int it = 0; it < 16; ++it) {
    const int rowQ = (it * 8 + wave) * 16 + l16;
    const bf16* qp = Qbase + (size_t)rowQ * CDK;
    v16bf a0 = ld_afrag(qp, base);
    v16bf a1 = ld_afrag(qp + 32, base);
#pragma unroll
    for (int cs = 0; cs < 4; ++cs) {
      const bf16* kp = &Ks[(cs * 16 + l16) * 72];
      v16bf b0 = ld_bfrag(kp, lhalf);
      v16bf b1 = ld_bfrag(kp + 32, lhalf);
      v8f sc;
#pragma unroll
      for (int r = 0; r < 8; ++r) sc[r] = 0.0f;
      sc = wmma_bf16(a0, b0, sc);
      sc = wmma_bf16(a1, b1, sc);
      float tmax = -3.0e38f;
#pragma unroll
      for (int r = 0; r < 8; ++r) {
        float t = sc[r] * 0.125f;  // 1/sqrt(64)
        sc[r] = t;
        tmax = fmaxf(tmax, t);
      }
      tmax = fmaxf(tmax, __shfl_xor(tmax, 16, 32));  // merge lane pair (col)
      float nm = fmaxf(mst[cs], tmax);
      float p = 0.0f;
#pragma unroll
      for (int r = 0; r < 8; ++r) p += __expf(sc[r] - nm);
      p += __shfl_xor(p, 16, 32);
      sst[cs] = sst[cs] * __expf(mst[cs] - nm) + p;
      mst[cs] = nm;
    }
  }
  if (lhalf == 0) {
#pragma unroll
    for (int cs = 0; cs < 4; ++cs) {
      lm[wave][cs * 16 + l16] = mst[cs];
      ls[wave][cs * 16 + l16] = sst[cs];
    }
  }
  __syncthreads();
  if (tid < 64) {  // merge 8 wave states (log-sum-exp merge)
    float M = -3.0e38f;
#pragma unroll
    for (int w = 0; w < 8; ++w) M = fmaxf(M, lm[w][tid]);
    float Ssum = 0.0f;
#pragma unroll
    for (int w = 0; w < 8; ++w) Ssum += ls[w][tid] * __expf(lm[w][tid] - M);
    m_arr[(size_t)bh * CS + s0 + tid] = M;
    r_arr[(size_t)bh * CS + s0 + tid] = 1.0f / Ssum;
  }
}

// ---------------------------------------------------------------------------
// Attention pass B: heads[q,v] = sum_s exp(score*scl - m[s]) * (V[s,v]/den[s])
// Block = 128 queries of one (b,h); s-loop in chunks of 32. Probability tile
// is staged through LDS to re-layout C-fragment -> A-fragment for 2nd WMMA.
// ---------------------------------------------------------------------------
__global__ __launch_bounds__(256) void attn_av_kernel(
    const bf16* __restrict__ Qb, const bf16* __restrict__ Kb,
    const bf16* __restrict__ Vb, const float* __restrict__ m_arr,
    const float* __restrict__ r_arr, bf16* __restrict__ headsb) {
  __shared__ bf16 Ks[32 * 72];
  __shared__ bf16 Vt[64 * 40];         // V' transposed: Vt[v][s]
  __shared__ bf16 Pt[8 * 16 * 40];     // per-wave prob tiles [16q x 32s]
  __shared__ float sm[32];

  const int tid = threadIdx.x;
  const int wave = tid >> 5, lane = tid & 31;
  const int l16 = lane & 15, lhalf = lane >> 4, base = lhalf * 8;
  const int bh = blockIdx.z * CH + blockIdx.y;
  const int q0 = blockIdx.x * 128;
  const bf16* Qbase = Qb + (size_t)bh * CS * CDK;
  const bf16* Kbase = Kb + (size_t)bh * CS * CDK;
  const bf16* Vbase = Vb + (size_t)bh * CS * CDK;
  const size_t statb = (size_t)bh * CS;

  v8f hacc[4];
#pragma unroll
  for (int vs = 0; vs < 4; ++vs)
#pragma unroll
    for (int r = 0; r < 8; ++r) hacc[vs][r] = 0.0f;

  for (int sci = 0; sci < CS / 32; ++sci) {
    const int s0 = sci * 32;
    __syncthreads();
    {  // K tile 32x64 -> LDS (async, row-major)
      int row = tid >> 3, off = (tid & 7) * 8;
      async_cp16(&Ks[row * 72 + off],
                 Kbase + (size_t)(s0 + row) * CDK + off);
    }
    if (tid < 32) sm[tid] = m_arr[statb + s0 + tid];
    {  // V' tile: scale by reciprocal denom, store transposed Vt[v][s]
      int s = tid >> 3, vc = (tid & 7) * 8;
      v8bf g = *(const v8bf*)(Vbase + (size_t)(s0 + s) * CDK + vc);
      float rr = r_arr[statb + s0 + s];
#pragma unroll
      for (int j = 0; j < 8; ++j)
        Vt[(vc + j) * 40 + s] = (bf16)((float)g[j] * rr);
    }
    wait_async0();
    __syncthreads();

    const int rowQ = q0 + wave * 16 + l16;
    const bf16* qp = Qbase + (size_t)rowQ * CDK;
    v16bf a0 = ld_afrag(qp, base);
    v16bf a1 = ld_afrag(qp + 32, base);
    bf16* P = &Pt[wave * 16 * 40];
#pragma unroll
    for (int st = 0; st < 2; ++st) {
      const bf16* kp = &Ks[(st * 16 + l16) * 72];
      v16bf b0 = ld_bfrag(kp, lhalf);
      v16bf b1 = ld_bfrag(kp + 32, lhalf);
      v8f s8;
#pragma unroll
      for (int r = 0; r < 8; ++r) s8[r] = 0.0f;
      s8 = wmma_bf16(a0, b0, s8);
      s8 = wmma_bf16(a1, b1, s8);
      float mc = sm[st * 16 + l16];
#pragma unroll
      for (int r = 0; r < 8; ++r)
        P[(r + 8 * lhalf) * 40 + st * 16 + l16] =
            (bf16)__expf(s8[r] * 0.125f - mc);
    }
    // LDS in-order per wave: safe to read the tile we just wrote.
    v16bf pa = ld_afrag(&P[l16 * 40], base);
#pragma unroll
    for (int vs = 0; vs < 4; ++vs) {
      v16bf bv = ld_bfrag(&Vt[(vs * 16 + l16) * 40], lhalf);
      hacc[vs] = wmma_bf16(pa, bv, hacc[vs]);
    }
  }

  const int b = blockIdx.z, h = blockIdx.y;
#pragma unroll
  for (int vs = 0; vs < 4; ++vs) {
#pragma unroll
    for (int r = 0; r < 8; ++r) {
      int q = q0 + wave * 16 + r + 8 * lhalf;
      size_t n = (size_t)b * CS + q;
      int col = h * CDK + vs * 16 + l16;
      headsb[n * (CH * CDK) + col] = (bf16)hacc[vs][r];
    }
  }
}

// ---------------------------------------------------------------------------
// Row LayerNorm over D=768; one block per row, 3 elems/thread.
// ---------------------------------------------------------------------------
__global__ __launch_bounds__(256) void layernorm_kernel(
    const float* __restrict__ in, const float* __restrict__ g,
    const float* __restrict__ bb, float* __restrict__ outF,
    bf16* __restrict__ outB) {
  __shared__ float red[16];
  const int row = blockIdx.x, tid = threadIdx.x;
  const float* x = in + (size_t)row * CD;
  float v[3], s = 0.0f, ss = 0.0f;
#pragma unroll
  for (int j = 0; j < 3; ++j) {
    v[j] = x[tid + j * 256];
    s += v[j];
    ss += v[j] * v[j];
  }
#pragma unroll
  for (int o = 16; o > 0; o >>= 1) {
    s += __shfl_xor(s, o, 32);
    ss += __shfl_xor(ss, o, 32);
  }
  const int wave = tid >> 5, lane = tid & 31;
  if (lane == 0) { red[wave] = s; red[8 + wave] = ss; }
  __syncthreads();
  float ts = 0.0f, tss = 0.0f;
#pragma unroll
  for (int w = 0; w < 8; ++w) { ts += red[w]; tss += red[8 + w]; }
  const float mean = ts * (1.0f / CD);
  const float var = tss * (1.0f / CD) - mean * mean;
  const float rstd = rsqrtf(var + 1e-5f);
#pragma unroll
  for (int j = 0; j < 3; ++j) {
    int c = tid + j * 256;
    float o = (v[j] - mean) * rstd * g[c] + bb[c];
    outF[(size_t)row * CD + c] = o;
    if (outB) outB[(size_t)row * CD + c] = (bf16)o;
  }
}

// ---------------------------------------------------------------------------
// Host-side orchestration.
// ---------------------------------------------------------------------------
extern "C" void kernel_launch(void* const* d_in, const int* in_sizes, int n_in,
                              void* d_out, int out_size, void* d_ws,
                              size_t ws_size, hipStream_t stream) {
  (void)in_sizes; (void)n_in; (void)out_size; (void)ws_size;
  const float* x     = (const float*)d_in[0];
  const float* w_q   = (const float*)d_in[1];
  const float* w_k   = (const float*)d_in[2];
  const float* w_v   = (const float*)d_in[3];
  const float* w_out = (const float*)d_in[4];
  const float* ln1_g = (const float*)d_in[5];
  const float* ln1_b = (const float*)d_in[6];
  const float* fc1_w = (const float*)d_in[7];
  const float* fc1_b = (const float*)d_in[8];
  const float* fc2_w = (const float*)d_in[9];
  const float* fc2_b = (const float*)d_in[10];
  const float* ln2_g = (const float*)d_in[11];
  const float* ln2_b = (const float*)d_in[12];

  char* ws = (char*)d_ws;
  size_t off = 0;
  auto alloc = [&](size_t bytes) {
    char* p = ws + off;
    off += (bytes + 255) & ~(size_t)255;
    return p;
  };
  bf16*  xb     = (bf16*)alloc((size_t)CN * CD * 2);        // also headsb later
  bf16*  Qb     = (bf16*)alloc((size_t)CB * CH * CS * CDK * 2);
  bf16*  Kb     = (bf16*)alloc((size_t)CB * CH * CS * CDK * 2);
  bf16*  Vb     = (bf16*)alloc((size_t)CB * CH * CS * CDK * 2);
  float* m_arr  = (float*)alloc((size_t)CB * CH * CS * 4);
  float* r_arr  = (float*)alloc((size_t)CB * CH * CS * 4);
  float* y1     = (float*)alloc((size_t)CN * CD * 4);        // also y2 later
  float* x1     = (float*)alloc((size_t)CN * CD * 4);
  bf16*  x1b    = (bf16*)alloc((size_t)CN * CD * 2);
  bf16*  h1b    = (bf16*)alloc((size_t)CN * CDFF * 2);
  bf16*  headsb = xb;  // xb dead after QKV projections
  float* y2     = y1;  // y1 dead after LN1

  // 1) x -> bf16
  cast_f32_bf16_kernel<<<(CN * CD) / 1024, 256, 0, stream>>>(x, xb);

  // 2) Q,K,V projections (head-major bf16 outputs)
  gemm_bf16_kernel<<<dim3(CN / 128, 1, CH), 256, 0, stream>>>(
      xb, CD, w_q, (long long)CD * CDK, CDK, CD,
      nullptr, nullptr, nullptr, Qb, CDK, 0, 1);
  gemm_bf16_kernel<<<dim3(CN / 128, 1, CH), 256, 0, stream>>>(
      xb, CD, w_k, (long long)CD * CDK, CDK, CD,
      nullptr, nullptr, nullptr, Kb, CDK, 0, 1);
  gemm_bf16_kernel<<<dim3(CN / 128, 1, CH), 256, 0, stream>>>(
      xb, CD, w_v, (long long)CD * CDK, CDK, CD,
      nullptr, nullptr, nullptr, Vb, CDK, 0, 1);

  // 3) column-softmax stats, then attn @ V'
  attn_stats_kernel<<<dim3(CS / 64, CH, CB), 256, 0, stream>>>(Qb, Kb, m_arr,
                                                               r_arr);
  attn_av_kernel<<<dim3(CS / 128, CH, CB), 256, 0, stream>>>(
      Qb, Kb, Vb, m_arr, r_arr, headsb);

  // 4) out-proj + residual(x) -> y1 (f32)
  gemm_bf16_kernel<<<dim3(CN / 128, CD / 64, 1), 256, 0, stream>>>(
      headsb, CD, w_out, 0, CD, CD,
      nullptr, x, y1, nullptr, CD, 0, 0);

  // 5) LN1 -> x1 (f32) + x1b (bf16)
  layernorm_kernel<<<CN, 256, 0, stream>>>(y1, ln1_g, ln1_b, x1, x1b);

  // 6) FC1 + bias + relu -> h1b (bf16)
  gemm_bf16_kernel<<<dim3(CN / 128, CDFF / 64, 1), 256, 0, stream>>>(
      x1b, CD, fc1_w, 0, CDFF, CD,
      fc1_b, nullptr, nullptr, h1b, CDFF, 1, 0);

  // 7) FC2 + bias + residual(x1) -> y2 (f32)
  gemm_bf16_kernel<<<dim3(CN / 128, CD / 64, 1), 256, 0, stream>>>(
      h1b, CDFF, fc2_w, 0, CD, CDFF,
      fc2_b, x1, y2, nullptr, CD, 0, 0);

  // 8) LN2 -> d_out (f32)
  layernorm_kernel<<<CN, 256, 0, stream>>>(y2, ln2_g, ln2_b, (float*)d_out,
                                           nullptr);
}
